// AutoregLSTM_69183333204607
// MI455X (gfx1250) — matrix operand order repaired and checked by reading, not compiled, not run on hardware
//
#include <hip/hip_runtime.h>

// ---------------- problem constants ----------------
constexpr int Bq   = 64;     // batch
constexpr int Tt   = 1024;   // time steps
constexpr int DIN  = 512;
constexpr int Hh   = 2048;
constexpr int DOUT = 512;
constexpr int KA   = DIN + DOUT + Hh;  // 3072  concat-K for gates GEMM
constexpr int NA   = 4 * Hh;           // 8192  gate columns
constexpr int KCH  = 128;              // K chunk per LDS stage

// output layout (floats): c_f | h_f | y_f | ys
constexpr size_t OUT_CF = 0;
constexpr size_t OUT_HF = (size_t)Bq * Hh;                 // 131072
constexpr size_t OUT_YF = OUT_HF + (size_t)Bq * Hh;        // 262144
constexpr size_t OUT_YS = OUT_YF + (size_t)Bq * DOUT;      // 294912

// ---------------- small helpers ----------------
typedef __attribute__((ext_vector_type(16))) __bf16 v16bf;
typedef __attribute__((ext_vector_type(8)))  float  v8f;

union Frag { uint4 q[2]; v16bf v; };

__device__ __forceinline__ unsigned short f32_to_bf16(float f) {
  unsigned int u = __float_as_uint(f);
  u += 0x7fffu + ((u >> 16) & 1u);       // round-to-nearest-even
  return (unsigned short)(u >> 16);
}
__device__ __forceinline__ float bf16_to_f32(unsigned short h) {
  return __uint_as_float(((unsigned int)h) << 16);
}
__device__ __forceinline__ float sigmoidf_(float x) {
  return 1.0f / (1.0f + __expf(-x));
}
__device__ __forceinline__ v8f v8f_zero() {
  v8f z;
#pragma unroll
  for (int i = 0; i < 8; ++i) z[i] = 0.0f;
  return z;
}

// LDS byte address = low 32 bits of generic pointer (flat LDS aperture rule)
__device__ __forceinline__ unsigned lds_addr(const void* p) {
  return (unsigned)(uintptr_t)p;
}
// Async 16-byte global -> LDS copy (ASYNCcnt tracked, bypasses VGPRs)
__device__ __forceinline__ void async_cp16(unsigned lds_dst, const void* gsrc) {
  asm volatile("global_load_async_to_lds_b128 %0, %1, off"
               :: "v"(lds_dst), "v"(gsrc) : "memory");
}
__device__ __forceinline__ void wait_async() {
  asm volatile("s_wait_asynccnt 0x0" ::: "memory");
}

// ---------------- one-time conversion kernels (write TRANSPOSED weights) ----
// WcatT[n][k] : n in [0,8192) gate columns, k in [0,3072)
__global__ void conv_wcat(const float* __restrict__ Wi, const float* __restrict__ Wh,
                          unsigned short* __restrict__ WcatT) {
  const size_t total = (size_t)KA * NA;
  for (size_t i = (size_t)blockIdx.x * blockDim.x + threadIdx.x; i < total;
       i += (size_t)gridDim.x * blockDim.x) {
    size_t r = i / NA, c = i % NA;
    float v = (r < (size_t)(DIN + DOUT)) ? Wi[r * NA + c] : Wh[(r - (DIN + DOUT)) * NA + c];
    WcatT[c * KA + r] = f32_to_bf16(v);
  }
}

// WoT[n][k] : n in [0,512), k in [0,2048)
__global__ void conv_wo(const float* __restrict__ Wo, unsigned short* __restrict__ WoT) {
  const size_t total = (size_t)Hh * DOUT;
  for (size_t i = (size_t)blockIdx.x * blockDim.x + threadIdx.x; i < total;
       i += (size_t)gridDim.x * blockDim.x) {
    size_t r = i / DOUT, c = i % DOUT;
    WoT[c * Hh + r] = f32_to_bf16(Wo[i]);
  }
}

// pack A_0 = [x_0 | y0 | h0] (bf16) and copy c0 -> c (fp32)
__global__ void init_state(const float* __restrict__ x, const float* __restrict__ y0,
                           const float* __restrict__ h0, const float* __restrict__ c0,
                           unsigned short* __restrict__ A0, float* __restrict__ c) {
  int i = blockIdx.x * blockDim.x + threadIdx.x;          // 0 .. 196607
  if (i < Bq * KA) {
    int b = i / KA, col = i % KA;
    float v;
    if (col < DIN)             v = x[(size_t)b * Tt * DIN + col];        // x[:,0,:]
    else if (col < DIN + DOUT) v = y0[b * DOUT + (col - DIN)];
    else                       v = h0[b * Hh + (col - DIN - DOUT)];
    A0[i] = f32_to_bf16(v);
  }
  if (i < Bq * Hh) c[i] = c0[i];
}

// ---------------- per-step kernel A: gates GEMM + pointwise ----------------
// grid = 128 blocks (gate-col tiles of 16), block = 128 threads = 4 waves.
// wave w computes M-tile w (rows 16w..16w+15) for all 4 gates on cols j0..j0+15.
// Double-buffered LDS, filled with async global->LDS copies.
__global__ __launch_bounds__(128) void lstm_gates(
    const unsigned short* __restrict__ WcatT,  // [8192][3072] bf16 (transposed)
    const unsigned short* __restrict__ Ain,    // [64][3072]  bf16 (step t)
    unsigned short* __restrict__ Anext,        // [64][3072]  bf16 (step t+1): h slice
    const float* __restrict__ bias,            // [8192]
    float* __restrict__ c,                     // [64][2048] fp32 state
    unsigned short* __restrict__ hbf)          // [64][2048] bf16
{
  __shared__ unsigned short lA[2][Bq * KCH];          // 2 x 16 KB
  __shared__ unsigned short lB[2][4 * 16 * KCH];      // 2 x 16 KB, [gate][n][k]

  const int tid   = threadIdx.x;
  const int lane  = tid & 31;
  const int wave  = tid >> 5;       // 0..3  -> M tile
  const int j0    = blockIdx.x * 16;
  const int khalf = lane >> 4;
  const int ln    = lane & 15;

  // staging geometry (per thread)
  const int arow = tid >> 1, ahalf = tid & 1;            // A: 64 rows x 256B
  const int ncol = tid >> 3, nseg = tid & 7;             // B: 16 cols x 8 segs(32B)

  auto stage = [&](int kc, int buf) {
    // A chunk: rows 0..63, k in [kc, kc+128): contiguous 256B per row
    {
      const unsigned short* g = Ain + (size_t)arow * KA + kc + ahalf * 64;
      unsigned d = lds_addr(&lA[buf][arow * KCH + ahalf * 64]);
#pragma unroll
      for (int i = 0; i < 4; ++i) async_cp16(d + 32 * i, g + 16 * i);
#pragma unroll
      for (int i = 0; i < 4; ++i) async_cp16(d + 32 * i + 16, g + 16 * i + 8);
    }
    // B chunk: 4 gates x 16 cols x 128 k, contiguous 256B per (gate,col)
#pragma unroll
    for (int g4 = 0; g4 < 4; ++g4) {
      const unsigned short* g =
          WcatT + (size_t)(g4 * Hh + j0 + ncol) * KA + kc + nseg * 16;
      unsigned d = lds_addr(&lB[buf][g4 * (16 * KCH) + ncol * KCH + nseg * 16]);
      async_cp16(d, g);
      async_cp16(d + 16, g + 8);
    }
  };

  v8f acc[4];
#pragma unroll
  for (int g = 0; g < 4; ++g) acc[g] = v8f_zero();

  stage(0, 0);
  wait_async();
  __syncthreads();

  int cur = 0;
  for (int kc = 0; kc < KA; kc += KCH) {
    const int nxt = cur ^ 1;
    if (kc + KCH < KA) stage(kc + KCH, nxt);       // prefetch next chunk (async)

    const unsigned short* cA = lA[cur];
    const unsigned short* cB = lB[cur];
#pragma unroll
    for (int ks = 0; ks < KCH; ks += 32) {
      Frag af;
      {
        const int m = wave * 16 + ln;
        const uint4* ap = (const uint4*)(&cA[m * KCH + ks + khalf * 8]);
        af.q[0] = ap[0];
        af.q[1] = ap[2];
      }
#pragma unroll
      for (int g = 0; g < 4; ++g) {
        Frag bf;
        const uint4* bp =
            (const uint4*)(&cB[g * (16 * KCH) + ln * KCH + ks + 8 * khalf]);
        bf.q[0] = bp[0];
        bf.q[1] = bp[2];
        acc[g] = __builtin_amdgcn_wmma_f32_16x16x32_bf16(
            false, af.v, false, bf.v, (short)0, acc[g], false, false);
      }
    }
    wait_async();        // our prefetch landed in LDS
    __syncthreads();     // everyone done reading cur / writing nxt
    cur = nxt;
  }

  // ---- bias + gate pointwise (i,f,g,o) ----
  const int col = j0 + ln;
  const float bi  = bias[0 * Hh + col];
  const float bff = bias[1 * Hh + col];
  const float bg  = bias[2 * Hh + col];
  const float bo_ = bias[3 * Hh + col];
#pragma unroll
  for (int e = 0; e < 8; ++e) {
    const int brow = wave * 16 + e + 8 * khalf;     // C/D layout: M = e + 8*(lane>=16)
    const float zi = acc[0][e] + bi;
    const float zf = acc[1][e] + bff;
    const float zg = acc[2][e] + bg;
    const float zo = acc[3][e] + bo_;
    const size_t idx = (size_t)brow * Hh + col;
    const float cp = c[idx];
    const float nc = sigmoidf_(zf) * cp + sigmoidf_(zi) * tanhf(zg);
    const float nh = sigmoidf_(zo) * tanhf(nc);
    c[idx] = nc;
    const unsigned short hb = f32_to_bf16(nh);
    hbf[idx] = hb;
    Anext[(size_t)brow * KA + (DIN + DOUT) + col] = hb;   // h slice of next A
  }
}

// ---------------- per-step kernel B: y = h@Wo + bo, plus x_{t+1} pack ----------------
// grid = 40 blocks: 0..31 GEMM (N tiles of 16), 32..39 pack x_{t+1}.
__global__ __launch_bounds__(128) void lstm_out(
    const unsigned short* __restrict__ WoT,   // [512][2048] bf16 (transposed)
    const unsigned short* __restrict__ hbf,   // [64][2048]  bf16
    const float* __restrict__ bo,             // [512]
    const float* __restrict__ x,              // [64][1024][512]
    unsigned short* __restrict__ Anext,       // [64][3072] (step t+1)
    float* __restrict__ ys,                   // [64][1024][512]
    int t)
{
  __shared__ unsigned short lA[2][Bq * KCH];   // 2 x 16 KB
  __shared__ unsigned short lB[2][16 * KCH];   // 2 x  4 KB  [n][k]

  if (blockIdx.x >= 32) {                      // ---- pack x_{t+1} ----
    if (t + 1 < Tt) {
      int i0 = (blockIdx.x - 32) * 128 + threadIdx.x;      // 0..1023
      for (int i = i0; i < Bq * DIN; i += 1024) {
        int b = i >> 9, col = i & 511;
        Anext[(size_t)b * KA + col] =
            f32_to_bf16(x[((size_t)b * Tt + (t + 1)) * DIN + col]);
      }
    }
    return;
  }

  const int tid   = threadIdx.x;
  const int lane  = tid & 31;
  const int wave  = tid >> 5;
  const int n0    = blockIdx.x * 16;
  const int khalf = lane >> 4;
  const int ln    = lane & 15;

  const int arow = tid >> 1, ahalf = tid & 1;
  const int ncol = tid >> 3, nseg = tid & 7;

  auto stage = [&](int kc, int buf) {
    {
      const unsigned short* g = hbf + (size_t)arow * Hh + kc + ahalf * 64;
      unsigned d = lds_addr(&lA[buf][arow * KCH + ahalf * 64]);
#pragma unroll
      for (int i = 0; i < 4; ++i) async_cp16(d + 32 * i, g + 16 * i);
#pragma unroll
      for (int i = 0; i < 4; ++i) async_cp16(d + 32 * i + 16, g + 16 * i + 8);
    }
    {
      const unsigned short* g = WoT + (size_t)(n0 + ncol) * Hh + kc + nseg * 16;
      unsigned d = lds_addr(&lB[buf][ncol * KCH + nseg * 16]);
      async_cp16(d, g);
      async_cp16(d + 16, g + 8);
    }
  };

  v8f acc = v8f_zero();

  stage(0, 0);
  wait_async();
  __syncthreads();

  int cur = 0;
  for (int kc = 0; kc < Hh; kc += KCH) {
    const int nxt = cur ^ 1;
    if (kc + KCH < Hh) stage(kc + KCH, nxt);

    const unsigned short* cA = lA[cur];
    const unsigned short* cB = lB[cur];
#pragma unroll
    for (int ks = 0; ks < KCH; ks += 32) {
      Frag af, bf;
      {
        const int m = wave * 16 + ln;
        const uint4* ap = (const uint4*)(&cA[m * KCH + ks + khalf * 8]);
        af.q[0] = ap[0];
        af.q[1] = ap[2];
      }
      {
        const uint4* bp = (const uint4*)(&cB[ln * KCH + ks + 8 * khalf]);
        bf.q[0] = bp[0];
        bf.q[1] = bp[2];
      }
      acc = __builtin_amdgcn_wmma_f32_16x16x32_bf16(
          false, af.v, false, bf.v, (short)0, acc, false, false);
    }
    wait_async();
    __syncthreads();
    cur = nxt;
  }

  const int col = n0 + ln;
  const float bb = bo[col];
#pragma unroll
  for (int e = 0; e < 8; ++e) {
    const int brow = wave * 16 + e + 8 * khalf;
    const float y = acc[e] + bb;
    ys[((size_t)brow * Tt + t) * DOUT + col] = y;
    Anext[(size_t)brow * KA + DIN + col] = f32_to_bf16(y);   // y slice of next A
  }
}

// ---------------- finalize: c_f, h_f, y_f ----------------
__global__ void finalize(const float* __restrict__ c, const unsigned short* __restrict__ hbf,
                         const float* __restrict__ ys, float* __restrict__ out) {
  int i = blockIdx.x * blockDim.x + threadIdx.x;        // 0..131071
  if (i < Bq * Hh) {
    out[OUT_CF + i] = c[i];
    out[OUT_HF + i] = bf16_to_f32(hbf[i]);
  }
  if (i < Bq * DOUT) {
    int b = i / DOUT, col = i % DOUT;
    out[OUT_YF + i] = ys[((size_t)b * Tt + (Tt - 1)) * DOUT + col];
  }
}

// ---------------- launch ----------------
extern "C" void kernel_launch(void* const* d_in, const int* in_sizes, int n_in,
                              void* d_out, int out_size, void* d_ws, size_t ws_size,
                              hipStream_t stream) {
  const float* x  = (const float*)d_in[0];
  const float* c0 = (const float*)d_in[1];
  const float* h0 = (const float*)d_in[2];
  const float* y0 = (const float*)d_in[3];
  const float* Wi = (const float*)d_in[4];
  const float* Wh = (const float*)d_in[5];
  const float* b  = (const float*)d_in[6];
  const float* Wo = (const float*)d_in[7];
  const float* bo = (const float*)d_in[8];
  float* out = (float*)d_out;

  // workspace layout (bytes)
  char* ws = (char*)d_ws;
  unsigned short* WcatT = (unsigned short*)(ws);                       // 50,331,648
  unsigned short* WoT   = (unsigned short*)(ws + 50331648u);           //  2,097,152
  unsigned short* Abuf0 = (unsigned short*)(ws + 52428800u);           //    393,216
  unsigned short* Abuf1 = (unsigned short*)(ws + 52822016u);           //    393,216
  float*          cbuf  = (float*)         (ws + 53215232u);           //    524,288
  unsigned short* hbf   = (unsigned short*)(ws + 53739520u);           //    262,144

  conv_wcat<<<2048, 256, 0, stream>>>(Wi, Wh, WcatT);
  conv_wo<<<512, 256, 0, stream>>>(Wo, WoT);
  init_state<<<768, 256, 0, stream>>>(x, y0, h0, c0, Abuf0, cbuf);

  float* ys = out + OUT_YS;
  for (int t = 0; t < Tt; ++t) {
    unsigned short* Ain   = (t & 1) ? Abuf1 : Abuf0;
    unsigned short* Anext = (t & 1) ? Abuf0 : Abuf1;
    lstm_gates<<<128, 128, 0, stream>>>(WcatT, Ain, Anext, b, cbuf, hbf);
    lstm_out<<<40, 128, 0, stream>>>(WoT, hbf, bo, x, Anext, ys, t);
  }

  finalize<<<512, 256, 0, stream>>>(cbuf, hbf, ys, out);
}
